// MHATokenNoLayerNorm_1125281432100
// MI455X (gfx1250) — matrix operand order repaired and checked by reading, not compiled, not run on hardware
//
#include <hip/hip_runtime.h>
#include <hip/hip_bf16.h>

typedef __attribute__((ext_vector_type(16))) __bf16 v16bf;
typedef __attribute__((ext_vector_type(8)))  __bf16 v8bf;
typedef __attribute__((ext_vector_type(8)))  float  v8f;
typedef __attribute__((ext_vector_type(4)))  unsigned v4u;
typedef __attribute__((ext_vector_type(8)))  int      v8i;
typedef __attribute__((ext_vector_type(4)))  int      v4i;

static __device__ __forceinline__ v16bf cat8(v8bf lo, v8bf hi) {
  return __builtin_shufflevector(lo, hi, 0, 1, 2, 3, 4, 5, 6, 7,
                                 8, 9, 10, 11, 12, 13, 14, 15);
}

// Low 32 bits of a generic pointer to __shared__ = LDS byte offset.
static __device__ __forceinline__ unsigned lds_addr_of(const void* p) {
  return (unsigned)(unsigned long long)(uintptr_t)p;
}

// ---------------------------------------------------------------------------
// TDM 2D tile load: global (bf16, row-major, row_stride elems) -> LDS with
// per-row padding (16-dword interval + 4-dword pad -> 80B LDS row stride).
// D# bitfields per CDNA5 ISA 8.3/8.4. Issued by one wave; EXEC ignored.
// 6-arg builtin variant (clang-23 / therock-10.0 headers).
// ---------------------------------------------------------------------------
static __device__ __forceinline__ void tdm_load_2d(unsigned lds_off,
                                                   const void* gaddr,
                                                   int tile_d0, int tile_d1,
                                                   unsigned long long stride0,
                                                   unsigned tensor_d0,
                                                   unsigned tensor_d1) {
  const unsigned long long ga = (unsigned long long)gaddr;
  v4u g0;
  g0[0] = 1u;                                     // count=1, user descriptor
  g0[1] = lds_off;                                // lds_addr (bytes)
  g0[2] = (unsigned)(ga & 0xffffffffu);           // global_addr[31:0]
  g0[3] = (unsigned)((ga >> 32) & 0x01ffffffu)    // global_addr[56:32]
          | (2u << 30);                           // type = 2 ("image")
  v8i g1;
  g1[0] = (int)((1u << 16)        // data_size = 2 bytes
                | (1u << 20)      // pad_enable
                | (3u << 22)      // pad_interval: 16 dwords (one 64B row)
                | (3u << 25));    // pad_amount: 4 dwords (16B)
  g1[1] = (int)((tensor_d0 & 0xffffu) << 16);                       // dim0 lo16
  g1[2] = (int)((tensor_d0 >> 16) | ((tensor_d1 & 0xffffu) << 16)); // dim0 hi / dim1 lo
  g1[3] = (int)((tensor_d1 >> 16) | ((unsigned)tile_d0 << 16));     // dim1 hi / tile_dim0
  g1[4] = (int)((unsigned)tile_d1 & 0xffffu);                       // tile_dim1 (tile_dim2=0)
  g1[5] = (int)(stride0 & 0xffffffffu);                             // dim0_stride lo32
  g1[6] = (int)((stride0 >> 32) & 0xffffu);                         // dim0_stride hi16
  g1[7] = 0;
  v4i z4 = {};
  v8i z8 = {};
  __builtin_amdgcn_tensor_load_to_lds(g0, g1, z4, z4, z8, 0);
}

// ---------------------------------------------------------------------------
// Pre-pass: fp32 -> bf16 elementwise (n multiple of 4).
// ---------------------------------------------------------------------------
__global__ __launch_bounds__(256) void cvt_bf16(const float* __restrict__ in,
                                                __bf16* __restrict__ out, int n) {
  const int i = (blockIdx.x * 256 + threadIdx.x) * 4;
  if (i < n) {
#pragma unroll
    for (int j = 0; j < 4; ++j) out[i + j] = (__bf16)in[i + j];
  }
}

// ---------------------------------------------------------------------------
// Pre-pass: W[768][768] fp32 -> WT[768][768] bf16 transposed (WT[n][k]=W[k][n]).
// ---------------------------------------------------------------------------
__global__ __launch_bounds__(256) void transpose_cvt(const float* __restrict__ W,
                                                     __bf16* __restrict__ WT) {
  __shared__ float tile[32][33];
  const int bn = blockIdx.x * 32;  // column (n) base
  const int bk = blockIdx.y * 32;  // row (k) base
  const int tx = threadIdx.x & 31;
  const int ty = threadIdx.x >> 5;  // 0..7
#pragma unroll
  for (int j = 0; j < 32; j += 8)
    tile[ty + j][tx] = W[(size_t)(bk + ty + j) * 768 + bn + tx];
  __syncthreads();
#pragma unroll
  for (int j = 0; j < 32; j += 8)
    WT[(size_t)(bn + ty + j) * 768 + bk + tx] = (__bf16)tile[tx][ty + j];
}

// ---------------------------------------------------------------------------
// 128x64-tile bf16 WMMA GEMM with TDM-staged, double-buffered LDS tiles:
//   out[M,768] = A[M,768] @ W[768,768] (+bias), W given TRANSPOSED (WT[n][k]).
// Block = 128 threads = 4 wave32; wave w owns rows [32w, 32w+32).
// Wave 0 issues 2 tensor_load_to_lds per k-step; all fragment reads are
// contiguous ds_load_b128 pairs.
// ---------------------------------------------------------------------------
template <bool HAS_BIAS, bool OUT_BF16>
__global__ __launch_bounds__(128) void gemm_wmma_tdm(const __bf16* __restrict__ A,
                                                     const __bf16* __restrict__ WT,
                                                     const float* __restrict__ bias,
                                                     void* __restrict__ outp) {
  constexpr int KD = 768, ND = 768, BM = 128, BN = 64, BK = 32, KT = KD / BK;
  constexpr int LS = 40;  // LDS row stride (bf16) = 80B, matches TDM pad config
  __shared__ __align__(16) __bf16 As[2][BM][LS];  // [buf][m][k]
  __shared__ __align__(16) __bf16 Ws[2][BN][LS];  // [buf][n][k]

  const int tid  = threadIdx.x;
  const int wave = tid >> 5;
  const int lane = tid & 31;
  const int m0   = blockIdx.y * BM;
  const int n0   = blockIdx.x * BN;

  v8f acc[2][4] = {};

  const int am = lane & 15;
  const int kh = (lane >> 4) * 8;
  const int bn = lane & 15;
  const int bk = (lane >> 4) * 8;

  // issue TDM loads for k-step kt into buffer buf (wave 0 only)
  auto issue = [&](int kt, int buf) {
    if (wave == 0) {
      tdm_load_2d(lds_addr_of(&As[buf][0][0]),
                  A + (size_t)m0 * KD + kt * BK, BK, BM,
                  (unsigned long long)KD, (unsigned)KD, 1u << 20);
      tdm_load_2d(lds_addr_of(&Ws[buf][0][0]),
                  WT + (size_t)n0 * KD + kt * BK, BK, BN,
                  (unsigned long long)KD, (unsigned)KD, 1u << 20);
    }
  };

  issue(0, 0);
  for (int kt = 0; kt < KT; ++kt) {
    const int buf = kt & 1;
    if (kt + 1 < KT) issue(kt + 1, buf ^ 1);
    if (wave == 0) {
      if (kt + 1 < KT) __builtin_amdgcn_s_wait_tensorcnt(2);
      else             __builtin_amdgcn_s_wait_tensorcnt(0);
    }
    __syncthreads();  // tile kt visible to all waves

    v16bf af[2];
#pragma unroll
    for (int mt = 0; mt < 2; ++mt) {
      const __bf16* rp = &As[buf][wave * 32 + mt * 16 + am][0];
      af[mt] = cat8(*(const v8bf*)(rp + kh), *(const v8bf*)(rp + 16 + kh));
    }
#pragma unroll
    for (int nt = 0; nt < 4; ++nt) {
      const __bf16* bp = &Ws[buf][nt * 16 + bn][bk];
      const v16bf bf = cat8(*(const v8bf*)(bp), *(const v8bf*)(bp + 16));
      acc[0][nt] = __builtin_amdgcn_wmma_f32_16x16x32_bf16(
          false, af[0], false, bf, (short)0, acc[0][nt], false, false);
      acc[1][nt] = __builtin_amdgcn_wmma_f32_16x16x32_bf16(
          false, af[1], false, bf, (short)0, acc[1][nt], false, false);
    }
    __syncthreads();  // all waves done with buf before TDM overwrites it
  }

  // ---- epilogue ----
  const int cn = lane & 15;
#pragma unroll
  for (int mt = 0; mt < 2; ++mt) {
    const int rbase = m0 + wave * 32 + mt * 16 + (lane >> 4) * 8;
#pragma unroll
    for (int nt = 0; nt < 4; ++nt) {
      const int col = n0 + nt * 16 + cn;
      const float bv = HAS_BIAS ? bias[col] : 0.0f;
#pragma unroll
      for (int g = 0; g < 8; ++g) {
        const float v = acc[mt][nt][g] + bv;
        if (OUT_BF16)
          ((__bf16*)outp)[(size_t)(rbase + g) * ND + col] = (__bf16)v;
        else
          ((float*)outp)[(size_t)(rbase + g) * ND + col] = v;
      }
    }
  }
}

// ---------------------------------------------------------------------------
// Attention core: one block per (b, s, h). 128 threads = 4 waves.
// scores = (q_h 64x64) @ (kv_h)^T * scale + mask*-1e4 ; softmax over t ;
// res_h = alpha @ kv_h. kv staged in both [t][d] and [d][t] layouts so both
// GEMMs read B fragments as contiguous ds_load_b128 pairs.
// ---------------------------------------------------------------------------
__global__ __launch_bounds__(128) void attn_kernel(const __bf16* __restrict__ qb,
                                                   const __bf16* __restrict__ kvb,
                                                   const float* __restrict__ mask,
                                                   __bf16* __restrict__ resb) {
  constexpr int HD = 768;
  constexpr int LS = 72;  // 144B row stride
  const int h = blockIdx.x, s = blockIdx.y, b = blockIdx.z;

  __shared__ __align__(16) __bf16 qs[64][LS];
  __shared__ __align__(16) __bf16 kvs[64][LS];
  __shared__ __align__(16) __bf16 kvt[64][LS];
  __shared__ __align__(16) __bf16 al[64][LS];
  __shared__ float sc[64][65];
  __shared__ float mk[64];

  const int tid  = threadIdx.x;
  const int wave = tid >> 5;
  const int lane = tid & 31;

  {
    const int r = tid >> 1;
    const int c = (tid & 1) * 32;
    const __bf16* qp = qb + (size_t)(b * 64 + r) * HD + h * 64 + c;
    const __bf16* kp = kvb + (size_t)((b * 64 + s) * 64 + r) * HD + h * 64 + c;
#pragma unroll
    for (int j = 0; j < 32; ++j) qs[r][c + j] = qp[j];
#pragma unroll
    for (int j = 0; j < 32; ++j) kvs[r][c + j] = kp[j];
#pragma unroll
    for (int j = 0; j < 32; ++j) kvt[c + j][r] = kp[j];
  }
  if (tid < 64) mk[tid] = mask[(size_t)(b * 64 + s) * 64 + tid] * -10000.0f;
  __syncthreads();

  const int am = lane & 15;
  const int kh = (lane >> 4) * 8;
  const int bn = lane & 15;
  const int bk = (lane >> 4) * 8;
  const int cn = lane & 15;
  const int rb = wave * 16 + (lane >> 4) * 8;

  // ---- scores = q @ kv^T ----
  v8f acc[4] = {};
#pragma unroll
  for (int ks = 0; ks < 64; ks += 32) {
    const __bf16* rp = &qs[wave * 16 + am][ks];
    const v16bf af = cat8(*(const v8bf*)(rp + kh), *(const v8bf*)(rp + 16 + kh));
#pragma unroll
    for (int nt = 0; nt < 4; ++nt) {
      const __bf16* bp = &kvs[nt * 16 + bn][ks + bk];
      const v16bf bf = cat8(*(const v8bf*)(bp), *(const v8bf*)(bp + 16));
      acc[nt] = __builtin_amdgcn_wmma_f32_16x16x32_bf16(
          false, af, false, bf, (short)0, acc[nt], false, false);
    }
  }
  const float scale = 0.125f;  // 1/sqrt(64)
#pragma unroll
  for (int nt = 0; nt < 4; ++nt)
#pragma unroll
    for (int g = 0; g < 8; ++g)
      sc[rb + g][nt * 16 + cn] = acc[nt][g] * scale + mk[nt * 16 + cn];
  __syncthreads();

  // ---- softmax over t ----
  if (tid < 64) {
    float mx = -3.4e38f;
#pragma unroll 4
    for (int t = 0; t < 64; ++t) mx = fmaxf(mx, sc[tid][t]);
    float sum = 0.0f;
#pragma unroll 4
    for (int t = 0; t < 64; ++t) {
      const float e = __expf(sc[tid][t] - mx);
      sc[tid][t] = e;
      sum += e;
    }
    const float inv = 1.0f / sum;
#pragma unroll 4
    for (int t = 0; t < 64; ++t) al[tid][t] = (__bf16)(sc[tid][t] * inv);
  }
  __syncthreads();

  // ---- res = alpha @ kv ----
  v8f oacc[4] = {};
#pragma unroll
  for (int ks = 0; ks < 64; ks += 32) {
    const __bf16* rp = &al[wave * 16 + am][ks];
    const v16bf af = cat8(*(const v8bf*)(rp + kh), *(const v8bf*)(rp + 16 + kh));
#pragma unroll
    for (int nt = 0; nt < 4; ++nt) {
      const __bf16* bp = &kvt[nt * 16 + bn][ks + bk];
      const v16bf bf = cat8(*(const v8bf*)(bp), *(const v8bf*)(bp + 16));
      oacc[nt] = __builtin_amdgcn_wmma_f32_16x16x32_bf16(
          false, af, false, bf, (short)0, oacc[nt], false, false);
    }
  }
#pragma unroll
  for (int nt = 0; nt < 4; ++nt)
#pragma unroll
    for (int g = 0; g < 8; ++g) {
      const int q = rb + g;
      const int d = nt * 16 + cn;
      resb[((size_t)(b * 64 + q) * 64 + s) * HD + h * 64 + d] = (__bf16)oacc[nt][g];
    }
}

// ---------------------------------------------------------------------------
extern "C" void kernel_launch(void* const* d_in, const int* in_sizes, int n_in,
                              void* d_out, int out_size, void* d_ws, size_t ws_size,
                              hipStream_t stream) {
  const float* ini_q = (const float*)d_in[0];  // (8, 64, 768)
  const float* ini_k = (const float*)d_in[1];  // (8, 64, 64, 768)
  const float* maskp = (const float*)d_in[2];  // (8, 64, 64)
  const float* Wq    = (const float*)d_in[3];  // (768, 768)
  const float* Wkv   = (const float*)d_in[4];  // (768, 768)
  const float* Wo    = (const float*)d_in[5];  // (768, 768)
  const float* bo    = (const float*)d_in[6];  // (768,)
  float* out = (float*)d_out;                  // (8, 64, 64, 768) fp32

  char* ws = (char*)d_ws;
  size_t off = 0;
  auto carve = [&](size_t bytes) {
    char* p = ws + off;
    off += (bytes + 255) & ~(size_t)255;
    return p;
  };
  __bf16* aqb  = (__bf16*)carve((size_t)512 * 768 * 2);    // bf16(ini_q)
  __bf16* akb  = (__bf16*)carve((size_t)32768 * 768 * 2);  // bf16(ini_k)
  __bf16* WqT  = (__bf16*)carve((size_t)768 * 768 * 2);    // bf16(Wq^T)
  __bf16* WkvT = (__bf16*)carve((size_t)768 * 768 * 2);    // bf16(Wkv^T)
  __bf16* WoT  = (__bf16*)carve((size_t)768 * 768 * 2);    // bf16(Wo^T)
  __bf16* qb   = (__bf16*)carve((size_t)512 * 768 * 2);    // q proj
  __bf16* kvb  = (__bf16*)carve((size_t)32768 * 768 * 2);  // kv proj
  __bf16* resb = (__bf16*)carve((size_t)32768 * 768 * 2);  // attention result

  // ---- pre-pass: bf16 conversion + weight transpose ----
  cvt_bf16<<<(512 * 768 / 4 + 255) / 256, 256, 0, stream>>>(ini_q, aqb, 512 * 768);
  cvt_bf16<<<(32768 * 768 / 4 + 255) / 256, 256, 0, stream>>>(ini_k, akb, 32768 * 768);
  transpose_cvt<<<dim3(24, 24), 256, 0, stream>>>(Wq, WqT);
  transpose_cvt<<<dim3(24, 24), 256, 0, stream>>>(Wkv, WkvT);
  transpose_cvt<<<dim3(24, 24), 256, 0, stream>>>(Wo, WoT);

  dim3 blk(128);
  // q = ini_q @ Wq  -> bf16
  gemm_wmma_tdm<false, true><<<dim3(12, 4), blk, 0, stream>>>(aqb, WqT, nullptr, qb);
  // kv = ini_k @ Wkv -> bf16
  gemm_wmma_tdm<false, true><<<dim3(12, 256), blk, 0, stream>>>(akb, WkvT, nullptr, kvb);
  // attention core per (b, s, h)
  attn_kernel<<<dim3(12, 64, 8), blk, 0, stream>>>(qb, kvb, maskp, resb);
  // out = res @ Wo + bo -> fp32
  gemm_wmma_tdm<true, false><<<dim3(12, 256), blk, 0, stream>>>(resb, WoT, bo, out);
}
